// Decoder_4552665333777
// MI455X (gfx1250) — matrix operand order repaired and checked by reading, not compiled, not run on hardware
//
#include <hip/hip_runtime.h>
#include <hip/hip_bf16.h>

typedef __attribute__((ext_vector_type(16))) __bf16 v16bf;
typedef __attribute__((ext_vector_type(8)))  __bf16 v8bf;
typedef __attribute__((ext_vector_type(8)))  float  v8f;

namespace {
constexpr int Bb = 128, Tt = 32, Ee = 512, Hh = 512, Vv = 10000, LFf = 49;
constexpr int VvPad = 10048;               // 157 * 64, zero-padded fc rows
constexpr int G4 = 4 * Hh;                 // 2048
constexpr int BM = 128, BN = 64, BK = 32;  // block tile
constexpr int SA = 40, SB = 40;            // padded LDS row strides (bf16 elems)
constexpr int NK = 512 / BK;               // K chunks (all GEMMs have K=512)
}

// ---- CDNA5 async global->LDS copy (ASYNCcnt-tracked, per-lane 16B DMA) ----
__device__ __forceinline__ void async_copy_b128(void* lds_ptr, const void* gptr) {
  unsigned l = (unsigned)(uintptr_t)lds_ptr;  // low 32 bits of flat LDS ptr = DS addr
  unsigned long long g = (unsigned long long)(uintptr_t)gptr;
  asm volatile("global_load_async_to_lds_b128 %0, %1, off"
               :: "v"(l), "v"(g) : "memory");
}

#if defined(__has_builtin)
#if __has_builtin(__builtin_amdgcn_s_wait_asynccnt)
#define WAIT_ASYNC(n) __builtin_amdgcn_s_wait_asynccnt(n)
#endif
#endif
#ifndef WAIT_ASYNC
#define WAIT_ASYNC(n) asm volatile("s_wait_asynccnt %0" :: "n"(n) : "memory")
#endif

__device__ __forceinline__ float sigmoidf_(float x) { return 1.0f / (1.0f + __expf(-x)); }

__device__ __forceinline__ v8f wmma_bf16(v16bf a, v16bf b, v8f c) {
  return __builtin_amdgcn_wmma_f32_16x16x32_bf16(false, a, false, b, (short)0, c,
                                                 false, false);
}

// A fragment (16x32 bf16): lane holds row M=lane&15, k-halves at (lane>>4)*8 and +16.
__device__ __forceinline__ v16bf frag_a(const __bf16* Al, int row, int lane) {
  const __bf16* p = Al + row * SA + ((lane >> 4) << 3);
  v8bf lo = *(const v8bf*)p;
  v8bf hi = *(const v8bf*)(p + 16);
  return __builtin_shufflevector(lo, hi, 0, 1, 2, 3, 4, 5, 6, 7, 8, 9, 10, 11, 12, 13, 14, 15);
}

// B fragment (32x16 bf16 from row-major W[N][K]): lane holds col N=lane&15,
// k-halves at (lane>>4)*8 and +16.
__device__ __forceinline__ v16bf frag_b(const __bf16* Bl, int row, int lane) {
  const __bf16* p = Bl + row * SB + ((lane >> 4) << 3);
  v8bf lo = *(const v8bf*)p;
  v8bf hi = *(const v8bf*)(p + 16);
  return __builtin_shufflevector(lo, hi, 0, 1, 2, 3, 4, 5, 6, 7, 8, 9, 10, 11, 12, 13, 14, 15);
}

// ---------------- prep kernels ----------------
__global__ void cvt_f32_bf16_k(const float* __restrict__ s, __bf16* __restrict__ d, int n8) {
  int i = blockIdx.x * blockDim.x + threadIdx.x;  // groups of 8 elements
  if (i < n8) {
    const float4* ps = (const float4*)s + (size_t)i * 2;
    float4 a = ps[0], b = ps[1];
    v8bf o;
    o[0] = (__bf16)a.x; o[1] = (__bf16)a.y; o[2] = (__bf16)a.z; o[3] = (__bf16)a.w;
    o[4] = (__bf16)b.x; o[5] = (__bf16)b.y; o[6] = (__bf16)b.z; o[7] = (__bf16)b.w;
    ((v8bf*)d)[i] = o;
  }
}

__global__ void zero_bf16_k(__bf16* __restrict__ d, int n8) {
  int i = blockIdx.x * blockDim.x + threadIdx.x;
  if (i < n8) {
    v8bf z;
#pragma unroll
    for (int q = 0; q < 8; ++q) z[q] = (__bf16)0.f;
    ((v8bf*)d)[i] = z;
  }
}

__global__ void bias_sum_k(const float* __restrict__ a, const float* __restrict__ b,
                           float* __restrict__ o, int n) {
  int i = blockIdx.x * blockDim.x + threadIdx.x;
  if (i < n) o[i] = a[i] + b[i];
}

__global__ void mean_feat_k(const float* __restrict__ f, __bf16* __restrict__ mf) {
  int i = blockIdx.x * blockDim.x + threadIdx.x;  // 128*512 threads
  int b = i >> 9, k = i & 511;
  const float* p = f + ((size_t)b * LFf) * 512 + k;
  float s = 0.f;
#pragma unroll 7
  for (int l = 0; l < LFf; ++l) s += p[(size_t)l * 512];
  mf[i] = (__bf16)(s * (1.0f / 49.0f));
}

// ---------------- h0/c0 = mf @ W^T + b ----------------
__global__ __launch_bounds__(256) void init_state_k(
    const __bf16* __restrict__ mf, const __bf16* __restrict__ whb,
    const __bf16* __restrict__ wcb, const float* __restrict__ hb,
    const float* __restrict__ cb, __bf16* __restrict__ hout, float* __restrict__ cout) {
  const int tid = threadIdx.x, lane = tid & 31, wave = tid >> 5;
  const int wm = wave & 3, wn = wave >> 2;
  const int which = blockIdx.y;
  const int blockN = blockIdx.x * BN;
  const __bf16* Wg = which ? wcb : whb;
  const float* bg = which ? cb : hb;

  __shared__ __bf16 Al[2][BM * SA];
  __shared__ __bf16 Bl[2][BN * SB];

  v8f acc[2][2];
#pragma unroll
  for (int in = 0; in < 2; ++in) {
    float bv = bg[blockN + wn * 32 + in * 16 + (lane & 15)];
#pragma unroll
    for (int im = 0; im < 2; ++im)
#pragma unroll
      for (int r = 0; r < 8; ++r) acc[im][in][r] = bv;
  }

  const int ra = tid >> 1, ca = (tid & 1) * 16;
  const int rb = tid >> 2, cbx = (tid & 3) * 8;

  auto issue = [&](int kc) {
    int st = kc & 1, k0 = kc * BK;
    async_copy_b128(&Al[st][ra * SA + ca],     mf + (size_t)ra * 512 + k0 + ca);
    async_copy_b128(&Al[st][ra * SA + ca + 8], mf + (size_t)ra * 512 + k0 + ca + 8);
    async_copy_b128(&Bl[st][rb * SB + cbx],    Wg + (size_t)(blockN + rb) * 512 + k0 + cbx);
  };

  issue(0);
  for (int kc = 0; kc < NK; ++kc) {
    int s = kc & 1;
    if (kc + 1 < NK) { issue(kc + 1); WAIT_ASYNC(3); } else { WAIT_ASYNC(0); }
    __syncthreads();
    v16bf af[2];
#pragma unroll
    for (int im = 0; im < 2; ++im)
      af[im] = frag_a(Al[s], wm * 32 + im * 16 + (lane & 15), lane);
#pragma unroll
    for (int in = 0; in < 2; ++in) {
      v16bf bfr = frag_b(Bl[s], wn * 32 + in * 16 + (lane & 15), lane);
#pragma unroll
      for (int im = 0; im < 2; ++im) acc[im][in] = wmma_bf16(af[im], bfr, acc[im][in]);
    }
    __syncthreads();
  }

#pragma unroll
  for (int im = 0; im < 2; ++im)
#pragma unroll
    for (int in = 0; in < 2; ++in) {
      int mb = wm * 32 + im * 16 + ((lane >> 4) << 3);
      int n = blockN + wn * 32 + in * 16 + (lane & 15);
#pragma unroll
      for (int r = 0; r < 8; ++r) {
        int m = mb + r;
        if (which == 0) hout[(size_t)m * Hh + n] = (__bf16)acc[im][in][r];
        else            cout[(size_t)m * Hh + n] = acc[im][in][r];
      }
    }
}

// ------- Xpre[t*B+b, :] = embedding[captions] @ W_ih^T + (b_ih+b_hh) -------
__global__ __launch_bounds__(256) void embed_gates_k(
    const int* __restrict__ cap, const __bf16* __restrict__ emb,
    const __bf16* __restrict__ wih, const float* __restrict__ bias,
    float* __restrict__ xpre) {
  const int tid = threadIdx.x, lane = tid & 31, wave = tid >> 5;
  const int wm = wave & 3, wn = wave >> 2;
  const int blockN = blockIdx.x * BN;
  const int blockM = blockIdx.y * BM;

  __shared__ __bf16 Al[2][BM * SA];
  __shared__ __bf16 Bl[2][BN * SB];

  const int ra = tid >> 1, ca = (tid & 1) * 16;
  const int rb = tid >> 2, cbx = (tid & 3) * 8;
  const int m_a = blockM + ra;  // m = t*128 + b
  const __bf16* arow = emb + (size_t)cap[(m_a & 127) * Tt + (m_a >> 7)] * Ee;
  const __bf16* brow = wih + (size_t)(blockN + rb) * Ee;

  v8f acc[2][2];
#pragma unroll
  for (int in = 0; in < 2; ++in) {
    float bv = bias[blockN + wn * 32 + in * 16 + (lane & 15)];
#pragma unroll
    for (int im = 0; im < 2; ++im)
#pragma unroll
      for (int r = 0; r < 8; ++r) acc[im][in][r] = bv;
  }

  auto issue = [&](int kc) {
    int st = kc & 1, k0 = kc * BK;
    async_copy_b128(&Al[st][ra * SA + ca],     arow + k0 + ca);
    async_copy_b128(&Al[st][ra * SA + ca + 8], arow + k0 + ca + 8);
    async_copy_b128(&Bl[st][rb * SB + cbx],    brow + k0 + cbx);
  };

  issue(0);
  for (int kc = 0; kc < NK; ++kc) {
    int s = kc & 1;
    if (kc + 1 < NK) { issue(kc + 1); WAIT_ASYNC(3); } else { WAIT_ASYNC(0); }
    __syncthreads();
    v16bf af[2];
#pragma unroll
    for (int im = 0; im < 2; ++im)
      af[im] = frag_a(Al[s], wm * 32 + im * 16 + (lane & 15), lane);
#pragma unroll
    for (int in = 0; in < 2; ++in) {
      v16bf bfr = frag_b(Bl[s], wn * 32 + in * 16 + (lane & 15), lane);
#pragma unroll
      for (int im = 0; im < 2; ++im) acc[im][in] = wmma_bf16(af[im], bfr, acc[im][in]);
    }
    __syncthreads();
  }

#pragma unroll
  for (int im = 0; im < 2; ++im)
#pragma unroll
    for (int in = 0; in < 2; ++in) {
      int mb = blockM + wm * 32 + im * 16 + ((lane >> 4) << 3);
      int n = blockN + wn * 32 + in * 16 + (lane & 15);
#pragma unroll
      for (int r = 0; r < 8; ++r)
        xpre[(size_t)(mb + r) * G4 + n] = acc[im][in][r];
    }
}

// ------- one LSTM time step: gates = Xpre[t] + h @ W_hh^T; fused cell -------
__global__ __launch_bounds__(256) void lstm_step_k(
    int t, const __bf16* __restrict__ hin, __bf16* __restrict__ hout,
    float* __restrict__ cbuf, const float* __restrict__ xpre,
    const __bf16* __restrict__ whh, __bf16* __restrict__ hall) {
  const int tid = threadIdx.x, lane = tid & 31, wave = tid >> 5;
  const int wm = wave & 3, wn = wave >> 2;
  const int j0 = blockIdx.x * BN;  // hidden units j0..j0+63, all 4 gates

  __shared__ __bf16 Al[2][BM * SA];
  __shared__ __bf16 Bl[2][4][BN * SB];

  const int ra = tid >> 1, ca = (tid & 1) * 16;
  const int rb = tid >> 2, cbx = (tid & 3) * 8;

  v8f acc[4][2][2];
#pragma unroll
  for (int g = 0; g < 4; ++g)
#pragma unroll
    for (int im = 0; im < 2; ++im) {
      int mb = wm * 32 + im * 16 + ((lane >> 4) << 3);
#pragma unroll
      for (int in = 0; in < 2; ++in) {
        int n = g * Hh + j0 + wn * 32 + in * 16 + (lane & 15);
        const float* xp = xpre + (size_t)(t * Bb + mb) * G4 + n;
#pragma unroll
        for (int r = 0; r < 8; ++r) acc[g][im][in][r] = xp[(size_t)r * G4];
      }
    }

  auto issue = [&](int kc) {
    int st = kc & 1, k0 = kc * BK;
    async_copy_b128(&Al[st][ra * SA + ca],     hin + (size_t)ra * Hh + k0 + ca);
    async_copy_b128(&Al[st][ra * SA + ca + 8], hin + (size_t)ra * Hh + k0 + ca + 8);
#pragma unroll
    for (int g = 0; g < 4; ++g)
      async_copy_b128(&Bl[st][g][rb * SB + cbx],
                      whh + (size_t)(g * Hh + j0 + rb) * Hh + k0 + cbx);
  };

  issue(0);
  for (int kc = 0; kc < NK; ++kc) {
    int s = kc & 1;
    if (kc + 1 < NK) { issue(kc + 1); WAIT_ASYNC(6); } else { WAIT_ASYNC(0); }
    __syncthreads();
    v16bf af[2];
#pragma unroll
    for (int im = 0; im < 2; ++im)
      af[im] = frag_a(Al[s], wm * 32 + im * 16 + (lane & 15), lane);
#pragma unroll
    for (int g = 0; g < 4; ++g)
#pragma unroll
      for (int in = 0; in < 2; ++in) {
        v16bf bfr = frag_b(Bl[s][g], wn * 32 + in * 16 + (lane & 15), lane);
#pragma unroll
        for (int im = 0; im < 2; ++im)
          acc[g][im][in] = wmma_bf16(af[im], bfr, acc[g][im][in]);
      }
    __syncthreads();
  }

  // LSTM cell: all four gates for (m, j) live in the same lane/register slot.
#pragma unroll
  for (int im = 0; im < 2; ++im) {
    int mb = wm * 32 + im * 16 + ((lane >> 4) << 3);
#pragma unroll
    for (int in = 0; in < 2; ++in) {
      int j = j0 + wn * 32 + in * 16 + (lane & 15);
#pragma unroll
      for (int r = 0; r < 8; ++r) {
        int m = mb + r;
        float iv = acc[0][im][in][r];
        float fv = acc[1][im][in][r];
        float gv = acc[2][im][in][r];
        float ov = acc[3][im][in][r];
        float cold = cbuf[(size_t)m * Hh + j];
        float cn = sigmoidf_(fv) * cold + sigmoidf_(iv) * tanhf(gv);
        float hn = sigmoidf_(ov) * tanhf(cn);
        cbuf[(size_t)m * Hh + j] = cn;
        __bf16 hb = (__bf16)hn;
        hout[(size_t)m * Hh + j] = hb;
        hall[((size_t)t * Bb + m) * Hh + j] = hb;
      }
    }
  }
}

// ------- preds = H_all @ fc_w^T + fc_b, masked, scattered to [B,T,V] -------
__global__ __launch_bounds__(256) void fc_out_k(
    const __bf16* __restrict__ hall, const __bf16* __restrict__ fcw,
    const float* __restrict__ fcb, const int* __restrict__ len,
    float* __restrict__ out) {
  const int tid = threadIdx.x, lane = tid & 31, wave = tid >> 5;
  const int wm = wave & 3, wn = wave >> 2;
  const int blockN = blockIdx.x * BN;
  const int blockM = blockIdx.y * BM;

  __shared__ __bf16 Al[2][BM * SA];
  __shared__ __bf16 Bl[2][BN * SB];

  const int ra = tid >> 1, ca = (tid & 1) * 16;
  const int rb = tid >> 2, cbx = (tid & 3) * 8;
  const __bf16* arow = hall + (size_t)(blockM + ra) * Hh;
  const __bf16* brow = fcw + (size_t)(blockN + rb) * Hh;  // fcw zero-padded to VvPad rows

  v8f acc[2][2];
#pragma unroll
  for (int in = 0; in < 2; ++in) {
    int n = blockN + wn * 32 + in * 16 + (lane & 15);
    float bv = (n < Vv) ? fcb[n] : 0.f;
#pragma unroll
    for (int im = 0; im < 2; ++im)
#pragma unroll
      for (int r = 0; r < 8; ++r) acc[im][in][r] = bv;
  }

  auto issue = [&](int kc) {
    int st = kc & 1, k0 = kc * BK;
    async_copy_b128(&Al[st][ra * SA + ca],     arow + k0 + ca);
    async_copy_b128(&Al[st][ra * SA + ca + 8], arow + k0 + ca + 8);
    async_copy_b128(&Bl[st][rb * SB + cbx],    brow + k0 + cbx);
  };

  issue(0);
  for (int kc = 0; kc < NK; ++kc) {
    int s = kc & 1;
    if (kc + 1 < NK) { issue(kc + 1); WAIT_ASYNC(3); } else { WAIT_ASYNC(0); }
    __syncthreads();
    v16bf af[2];
#pragma unroll
    for (int im = 0; im < 2; ++im)
      af[im] = frag_a(Al[s], wm * 32 + im * 16 + (lane & 15), lane);
#pragma unroll
    for (int in = 0; in < 2; ++in) {
      v16bf bfr = frag_b(Bl[s], wn * 32 + in * 16 + (lane & 15), lane);
#pragma unroll
      for (int im = 0; im < 2; ++im) acc[im][in] = wmma_bf16(af[im], bfr, acc[im][in]);
    }
    __syncthreads();
  }

#pragma unroll
  for (int im = 0; im < 2; ++im)
#pragma unroll
    for (int in = 0; in < 2; ++in) {
      int mb = blockM + wm * 32 + im * 16 + ((lane >> 4) << 3);
      int n = blockN + wn * 32 + in * 16 + (lane & 15);
      if (n < Vv) {
#pragma unroll
        for (int r = 0; r < 8; ++r) {
          int m = mb + r;
          int tt = m >> 7, bb = m & 127;
          float v = (tt < len[bb]) ? acc[im][in][r] : 0.f;
          out[((size_t)bb * Tt + tt) * Vv + n] = v;
        }
      }
    }
}

extern "C" void kernel_launch(void* const* d_in, const int* in_sizes, int n_in,
                              void* d_out, int out_size, void* d_ws, size_t ws_size,
                              hipStream_t stream) {
  (void)in_sizes; (void)n_in; (void)out_size; (void)ws_size;
  const float* features  = (const float*)d_in[0];
  const int*   captions  = (const int*)d_in[1];
  const int*   lengths   = (const int*)d_in[2];
  const float* embedding = (const float*)d_in[3];
  const float* W_ih = (const float*)d_in[4];
  const float* W_hh = (const float*)d_in[5];
  const float* b_ih = (const float*)d_in[6];
  const float* b_hh = (const float*)d_in[7];
  const float* fc_w = (const float*)d_in[8];
  const float* fc_b = (const float*)d_in[9];
  const float* ihw  = (const float*)d_in[10];
  const float* ihb  = (const float*)d_in[11];
  const float* icw  = (const float*)d_in[12];
  const float* icb  = (const float*)d_in[13];
  float* out = (float*)d_out;

  char* ws = (char*)d_ws;
  size_t off = 0;
  auto alloc = [&](size_t bytes) -> void* {
    void* p = (void*)(ws + off);
    off += (bytes + 255) & ~(size_t)255;
    return p;
  };
  __bf16* wih_bf = (__bf16*)alloc((size_t)G4 * 512 * 2);
  __bf16* whh_bf = (__bf16*)alloc((size_t)G4 * 512 * 2);
  __bf16* fcw_bf = (__bf16*)alloc((size_t)VvPad * 512 * 2);  // zero-padded rows
  __bf16* emb_bf = (__bf16*)alloc((size_t)Vv * 512 * 2);
  __bf16* ihw_bf = (__bf16*)alloc((size_t)512 * 512 * 2);
  __bf16* icw_bf = (__bf16*)alloc((size_t)512 * 512 * 2);
  float*  biasv  = (float*)alloc((size_t)G4 * 4);
  __bf16* mf     = (__bf16*)alloc((size_t)Bb * 512 * 2);
  __bf16* hbuf0  = (__bf16*)alloc((size_t)Bb * Hh * 2);
  __bf16* hbuf1  = (__bf16*)alloc((size_t)Bb * Hh * 2);
  float*  cbuf   = (float*)alloc((size_t)Bb * Hh * 4);
  float*  xpre   = (float*)alloc((size_t)Tt * Bb * G4 * 4);
  __bf16* hall   = (__bf16*)alloc((size_t)Tt * Bb * Hh * 2);

  const int TPB = 256;
  auto cvt = [&](const float* s, __bf16* d, int n) {  // n multiple of 8
    int n8 = n / 8;
    cvt_f32_bf16_k<<<(n8 + TPB - 1) / TPB, TPB, 0, stream>>>(s, d, n8);
  };
  cvt(W_ih, wih_bf, G4 * 512);
  cvt(W_hh, whh_bf, G4 * 512);
  cvt(fc_w, fcw_bf, Vv * 512);
  cvt(embedding, emb_bf, Vv * 512);
  cvt(ihw, ihw_bf, 512 * 512);
  cvt(icw, icw_bf, 512 * 512);
  {
    int n8 = (VvPad - Vv) * 512 / 8;  // zero the fc weight pad rows
    zero_bf16_k<<<(n8 + TPB - 1) / TPB, TPB, 0, stream>>>(fcw_bf + (size_t)Vv * 512, n8);
  }
  bias_sum_k<<<(G4 + TPB - 1) / TPB, TPB, 0, stream>>>(b_ih, b_hh, biasv, G4);
  mean_feat_k<<<(Bb * 512) / TPB, TPB, 0, stream>>>(features, mf);

  init_state_k<<<dim3(512 / BN, 2), TPB, 0, stream>>>(mf, ihw_bf, icw_bf, ihb, icb,
                                                      hbuf0, cbuf);
  embed_gates_k<<<dim3(G4 / BN, (Tt * Bb) / BM), TPB, 0, stream>>>(captions, emb_bf,
                                                                   wih_bf, biasv, xpre);
  for (int t = 0; t < Tt; ++t) {
    const __bf16* hi = (t & 1) ? hbuf1 : hbuf0;
    __bf16* ho = (t & 1) ? hbuf0 : hbuf1;
    lstm_step_k<<<Hh / BN, TPB, 0, stream>>>(t, hi, ho, cbuf, xpre, whh_bf, hall);
  }
  fc_out_k<<<dim3(VvPad / BN, (Tt * Bb) / BM), TPB, 0, stream>>>(hall, fcw_bf,
                                                                 fc_b, lengths, out);
}
